// ImplicitMeshDecoder_20813411516643
// MI455X (gfx1250) — matrix operand order repaired and checked by reading, not compiled, not run on hardware
//
#include <hip/hip_runtime.h>

typedef __attribute__((ext_vector_type(16))) _Float16 v16h;
typedef __attribute__((ext_vector_type(8)))  _Float16 v8h;
typedef __attribute__((ext_vector_type(8)))  float    v8f;

#define HID        512
#define DLAT       512
#define GRID_RES   64
#define P_PTS      262144              // 64^3
#define NROWS      (2 * P_PTS)         // B * P = 524288
#define M_TILE     128                 // rows per workgroup
#define LDS_STRIDE 520                 // halfs per row (pad: avoids 64-bank conflicts, keeps 16B align)
#define BOUNDF     1.2f

__device__ __forceinline__ float silu_f(float x) {
    return x * (1.0f / (1.0f + __expf(-x)));
}

// ---------------------------------------------------------------------------
// Prep 1: transpose + fp32->fp16 convert W1, W2 into column-major [n][k]
// so each WMMA B-fragment is 16 contiguous halfs (32B) per lane.
// ---------------------------------------------------------------------------
__global__ void imd_prep_weights(const float* __restrict__ W1,
                                 const float* __restrict__ W2,
                                 _Float16* __restrict__ w1t,
                                 _Float16* __restrict__ w2t) {
    int id  = blockIdx.x * blockDim.x + threadIdx.x;   // 0 .. 2*512*512-1
    int m   = id >> 18;                                // which matrix
    int rem = id & (HID * HID - 1);
    int k   = rem >> 9;
    int n   = rem & (HID - 1);
    const float* src = m ? W2 : W1;
    _Float16*    dst = m ? w2t : w1t;
    dst[n * HID + k] = (_Float16)src[k * HID + n];
}

// ---------------------------------------------------------------------------
// Prep 2: lat0[b][n] = cad_latent[b] @ W0[:512, n] + b0[n]   (2x512 outputs)
// ---------------------------------------------------------------------------
__global__ void imd_prep_lat0(const float* __restrict__ lat,
                              const float* __restrict__ W0,
                              const float* __restrict__ b0,
                              float* __restrict__ lat0) {
    int id = blockIdx.x * blockDim.x + threadIdx.x;    // 0..1023
    int b  = id >> 9;
    int n  = id & (HID - 1);
    float acc = b0[n];
    for (int d = 0; d < DLAT; ++d)
        acc += lat[b * DLAT + d] * W0[d * HID + n];
    lat0[id] = acc;
}

// ---------------------------------------------------------------------------
// One fused GEMM+bias+SiLU layer: out[128][512] = silu(in[128][512] @ W + b)
// Each wave computes a 64x64 block: 4x4 grid of 16x16 WMMA tiles.
//   A frags from LDS (ds_load_b128), B frags from global f16 weights (L2-hot).
// ---------------------------------------------------------------------------
__device__ __forceinline__ void imd_gemm_layer(const _Float16* __restrict__ bufIn,
                                               const _Float16* __restrict__ wt,   // [n][k] f16
                                               const float* __restrict__ bias,
                                               _Float16* __restrict__ bufOut,
                                               int tid) {
    const int wave = tid >> 5;
    const int lane = tid & 31;
    const int hi   = lane >> 4;        // 0: lanes 0-15, 1: lanes 16-31
    const int l15  = lane & 15;
    const int akb  = hi << 4;          // A frag K sub-offset: 0 or 16
    const int bkb  = hi << 4;          // B frag K sub-offset: 0 or 16

    // 16 blocks of 64x64 over the 128x512 output; 8 waves -> 2 blocks each
    for (int s = 0; s < 2; ++s) {
        const int bb    = wave + (s << 3);   // 0..15
        const int mbase = (bb & 1) << 6;     // 0 or 64
        const int nbase = (bb >> 1) << 6;    // 0,64,...,448

        v8f acc[4][4];
#pragma unroll
        for (int i = 0; i < 4; ++i)
#pragma unroll
            for (int j = 0; j < 4; ++j)
#pragma unroll
                for (int r = 0; r < 8; ++r) acc[i][j][r] = 0.0f;

        for (int k0 = 0; k0 < HID; k0 += 32) {
            // B fragments: 32x16 f16. lane = (K half, N=l15); K = bkb + [0,16)
            v16h bf[4];
#pragma unroll
            for (int j = 0; j < 4; ++j) {
                const _Float16* p = wt + (nbase + (j << 4) + l15) * HID + k0 + bkb;
                bf[j] = *(const v16h*)p;                       // 32B contiguous
            }
            // A fragments: 16x32 f16. lane row = l15, K runs [akb,akb+16)
            v16h af[4];
#pragma unroll
            for (int i = 0; i < 4; ++i) {
                const _Float16* p = bufIn + (mbase + (i << 4) + l15) * LDS_STRIDE + k0 + akb;
                union { v16h v; v8h h[2]; } u;
                u.h[0] = *(const v8h*)(p);
                u.h[1] = *(const v8h*)(p + 8);
                af[i] = u.v;
            }
#pragma unroll
            for (int i = 0; i < 4; ++i)
#pragma unroll
                for (int j = 0; j < 4; ++j)
                    acc[i][j] = __builtin_amdgcn_wmma_f32_16x16x32_f16(
                        false, af[i], false, bf[j], (short)0, acc[i][j],
                        false, false);
        }

        // Epilogue: bias + SiLU + f16 store to the other LDS buffer.
        // C/D layout: VGPR r -> row r (lanes 0-15) or row r+8 (lanes 16-31); col = l15.
        const int roff = hi << 3;
#pragma unroll
        for (int j = 0; j < 4; ++j) {
            const int ncol = nbase + (j << 4) + l15;
            const float bj = bias[ncol];
#pragma unroll
            for (int i = 0; i < 4; ++i) {
#pragma unroll
                for (int r = 0; r < 8; ++r) {
                    float v = acc[i][j][r] + bj;
                    bufOut[(mbase + (i << 4) + roff + r) * LDS_STRIDE + ncol] =
                        (_Float16)silu_f(v);
                }
            }
        }
    }
}

// ---------------------------------------------------------------------------
// Main fused kernel: one WG = 128 rows through all 4 layers.
// ---------------------------------------------------------------------------
__global__ __launch_bounds__(256) void imd_decoder_main(
    const float* __restrict__ W0,        // (515,512) row-major; rows 512..514 = coord weights
    const float* __restrict__ b1,
    const float* __restrict__ b2,
    const float* __restrict__ W3,        // (512,1)
    const float* __restrict__ b3,
    const _Float16* __restrict__ w1t,    // (512,512) f16, [n][k]
    const _Float16* __restrict__ w2t,
    const float* __restrict__ lat0,      // (2,512): latent@W0 + b0
    float* __restrict__ out)             // (NROWS,) fp32
{
    extern __shared__ _Float16 smem[];
    _Float16* bufA = smem;
    _Float16* bufB = smem + M_TILE * LDS_STRIDE;

    const int tid  = threadIdx.x;
    const int row0 = blockIdx.x * M_TILE;

    // ---- Layer 0: h0 = silu(lat0[b] + x*W0c0 + y*W0c1 + z*W0c2) -> bufA (f16)
    const float step = (2.0f * BOUNDF) / (float)(GRID_RES - 1);
    for (int i = tid; i < M_TILE * HID; i += 256) {
        int rl = i >> 9;
        int n  = i & (HID - 1);
        int rg = row0 + rl;
        int b  = rg >> 18;               // P = 2^18
        int p  = rg & (P_PTS - 1);
        float x = -BOUNDF + step * (float)(p >> 12);
        float y = -BOUNDF + step * (float)((p >> 6) & 63);
        float z = -BOUNDF + step * (float)(p & 63);
        float v = lat0[b * HID + n]
                + x * W0[(DLAT + 0) * HID + n]
                + y * W0[(DLAT + 1) * HID + n]
                + z * W0[(DLAT + 2) * HID + n];
        bufA[rl * LDS_STRIDE + n] = (_Float16)silu_f(v);
    }
    __syncthreads();

    // ---- Layers 1 and 2 (WMMA GEMM + bias + SiLU), ping-pong LDS buffers
    imd_gemm_layer(bufA, w1t, b1, bufB, tid);
    __syncthreads();
    imd_gemm_layer(bufB, w2t, b2, bufA, tid);
    __syncthreads();

    // ---- Layer 3: sdf = h2 @ W3 + b3   (one row per thread, vectorized f16 reads)
    if (tid < M_TILE) {
        const _Float16* hrow = bufA + tid * LDS_STRIDE;
        float acc = 0.0f;
        for (int k = 0; k < HID; k += 8) {
            v8h hv = *(const v8h*)(hrow + k);
#pragma unroll
            for (int r = 0; r < 8; ++r)
                acc += (float)hv[r] * W3[k + r];
        }
        out[row0 + tid] = acc + b3[0];
    }
}

// ---------------------------------------------------------------------------
extern "C" void kernel_launch(void* const* d_in, const int* in_sizes, int n_in,
                              void* d_out, int out_size, void* d_ws, size_t ws_size,
                              hipStream_t stream) {
    const float* cad = (const float*)d_in[0];
    const float* W0  = (const float*)d_in[1];
    const float* b0  = (const float*)d_in[2];
    const float* W1  = (const float*)d_in[3];
    const float* b1  = (const float*)d_in[4];
    const float* W2  = (const float*)d_in[5];
    const float* b2  = (const float*)d_in[6];
    const float* W3  = (const float*)d_in[7];
    const float* b3  = (const float*)d_in[8];
    float* out = (float*)d_out;

    char* ws = (char*)d_ws;
    _Float16* w1t  = (_Float16*)ws;                    // 512 KB
    _Float16* w2t  = (_Float16*)(ws + (512 << 10));    // 512 KB
    float*    lat0 = (float*)(ws + (1024 << 10));      // 4 KB

    imd_prep_weights<<<(2 * HID * HID) / 256, 256, 0, stream>>>(W1, W2, w1t, w2t);
    imd_prep_lat0<<<4, 256, 0, stream>>>(cad, W0, b0, lat0);

    const int lds_bytes = 2 * M_TILE * LDS_STRIDE * (int)sizeof(_Float16); // ~266 KB (CDNA5: 320 KB/WG)
    hipFuncSetAttribute((const void*)imd_decoder_main,
                        hipFuncAttributeMaxDynamicSharedMemorySize, lds_bytes);
    imd_decoder_main<<<NROWS / M_TILE, 256, lds_bytes, stream>>>(
        W0, b1, b2, W3, b3, w1t, w2t, lat0, out);
}
